// LARNNCell_27058293965418
// MI455X (gfx1250) — matrix operand order repaired and checked by reading, not compiled, not run on hardware
//
#include <hip/hip_runtime.h>
#include <hip/hip_bf16.h>
#include <math.h>

typedef __bf16 bf16_t;
typedef bf16_t v16bf __attribute__((ext_vector_type(16)));
typedef bf16_t v8bf  __attribute__((ext_vector_type(8)));
typedef float  v8f   __attribute__((ext_vector_type(8)));

#define EPSC 1e-5f
#define B_    512
#define IN_   512
#define H_    1024
#define HEADS_ 16
#define W_    32
#define P_    64
#define DK_   64          // H/HEADS
#define KQ_   (IN_ + H_)  // 1536
#define KKV_  (H_ + P_)   // 1088
#define KCAT_ (IN_ + H_ + H_) // 2560
#define N4H_  (4 * H_)    // 4096

// ---------------------------------------------------------------- converts
__global__ void f32_to_bf16_k(const float* __restrict__ in, bf16_t* __restrict__ out, size_t n) {
    size_t i = (size_t)blockIdx.x * blockDim.x + threadIdx.x;
    if (i < n) out[i] = (bf16_t)in[i];
}

// xha[b, 0:512]=x, [512:1536]=prev_hidden  (cols 1536.. filled later by attn GEMM epilogue)
__global__ void build_xha_k(const float* __restrict__ x, const float* __restrict__ ph,
                            bf16_t* __restrict__ xha) {
    size_t i = (size_t)blockIdx.x * blockDim.x + threadIdx.x;
    size_t n = (size_t)B_ * KQ_;
    if (i >= n) return;
    int row = (int)(i / KQ_), c = (int)(i % KQ_);
    float v = (c < IN_) ? x[(size_t)row * IN_ + c] : ph[(size_t)row * H_ + (c - IN_)];
    xha[(size_t)row * KCAT_ + c] = (bf16_t)v;
}

// Wcat[n, 0:512]=W_ih, [512:1536]=W_hh, [1536:2560]=W_ac
__global__ void build_wcat_k(const float* __restrict__ wih, const float* __restrict__ whh,
                             const float* __restrict__ wac, bf16_t* __restrict__ wc) {
    size_t i = (size_t)blockIdx.x * blockDim.x + threadIdx.x;
    size_t n = (size_t)N4H_ * KCAT_;
    if (i >= n) return;
    int row = (int)(i / KCAT_), c = (int)(i % KCAT_);
    float v;
    if (c < IN_)            v = wih[(size_t)row * IN_ + c];
    else if (c < IN_ + H_)  v = whh[(size_t)row * H_ + (c - IN_)];
    else                    v = wac[(size_t)row * H_ + (c - IN_ - H_)];
    wc[i] = (bf16_t)v;
}

// kv[b*W+w, 0:1024] = past_cells[w,b,:], [1024:1088] = PE(w,:)
__global__ void build_kv_k(const float* __restrict__ pcs, bf16_t* __restrict__ kv) {
    size_t i = (size_t)blockIdx.x * blockDim.x + threadIdx.x;
    size_t n = (size_t)B_ * W_ * KKV_;
    if (i >= n) return;
    int row = (int)(i / KKV_), c = (int)(i % KKV_);
    int b = row / W_, w = row % W_;
    float v;
    if (c < H_) {
        v = pcs[((size_t)w * B_ + b) * H_ + c];
    } else {
        int cp = c - H_;                 // 0..63
        int j  = cp >> 1;                // pair index
        float div = expf(-(float)(2 * j) * (logf(10000.0f) / (float)P_));
        float a = (float)w * div;
        v = (cp & 1) ? cosf(a) : sinf(a);
    }
    kv[i] = (bf16_t)v;
}

// ---------------------------------------------------------------- bf16 WMMA GEMM
// C[M,N] = A[M,K] @ Wt[N,K]^T  (+bias +bias2, optional ELU, optional BN), M,N %64==0, K %32==0
__global__ __launch_bounds__(128)
void gemm_bf16_wmma(const bf16_t* __restrict__ A, int lda,
                    const bf16_t* __restrict__ Wt, int ldw,
                    const float* __restrict__ bias, const float* __restrict__ bias2,
                    int M, int N, int K, int elu,
                    const float* __restrict__ bng, const float* __restrict__ bnb,
                    const float* __restrict__ bnm, const float* __restrict__ bnv,
                    float* __restrict__ outF, int ldf,
                    bf16_t* __restrict__ outH, int ldh, int colOff) {
    const int lane  = threadIdx.x & 31;
    const int wave  = threadIdx.x >> 5;
    const int tileM = blockIdx.y * 64 + wave * 16;
    const int tileN = blockIdx.x * 64;
    if (tileM >= M) return;

    const int mrow = tileM + (lane & 15);
    const int khA  = (lane >= 16) ? 16 : 0;   // A: K 0-15 / 16-31 per lane group
    const int nrow = tileN + (lane & 15);
    const int kB   = (lane >= 16) ? 16 : 0;   // B: K 0-15 / 16-31 per lane group

    const bf16_t* aRow  = A  + (size_t)mrow * lda + khA;
    const bf16_t* bRow0 = Wt + (size_t)(nrow +  0) * ldw + kB;
    const bf16_t* bRow1 = Wt + (size_t)(nrow + 16) * ldw + kB;
    const bf16_t* bRow2 = Wt + (size_t)(nrow + 32) * ldw + kB;
    const bf16_t* bRow3 = Wt + (size_t)(nrow + 48) * ldw + kB;

    v8f c0 = {}, c1 = {}, c2 = {}, c3 = {};

    for (int k0 = 0; k0 < K; k0 += 32) {
        // A fragment: two contiguous 16B runs
        v8bf alo = *(const v8bf*)(aRow + k0);
        v8bf ahi = *(const v8bf*)(aRow + k0 + 8);
        v16bf a = __builtin_shufflevector(alo, ahi, 0,1,2,3,4,5,6,7,8,9,10,11,12,13,14,15);
        __builtin_prefetch(aRow + k0 + 32, 0, 1);   // global_prefetch_b8 next K-slab

        // B fragments: one contiguous 32B run each
        v8bf t0 = *(const v8bf*)(bRow0 + k0); v8bf t1 = *(const v8bf*)(bRow0 + k0 + 8);
        v16bf b = __builtin_shufflevector(t0, t1, 0,1,2,3,4,5,6,7,8,9,10,11,12,13,14,15);
        c0 = __builtin_amdgcn_wmma_f32_16x16x32_bf16(false, a, false, b, (short)0, c0, false, false);

        t0 = *(const v8bf*)(bRow1 + k0); t1 = *(const v8bf*)(bRow1 + k0 + 8);
        b = __builtin_shufflevector(t0, t1, 0,1,2,3,4,5,6,7,8,9,10,11,12,13,14,15);
        c1 = __builtin_amdgcn_wmma_f32_16x16x32_bf16(false, a, false, b, (short)0, c1, false, false);

        t0 = *(const v8bf*)(bRow2 + k0); t1 = *(const v8bf*)(bRow2 + k0 + 8);
        b = __builtin_shufflevector(t0, t1, 0,1,2,3,4,5,6,7,8,9,10,11,12,13,14,15);
        c2 = __builtin_amdgcn_wmma_f32_16x16x32_bf16(false, a, false, b, (short)0, c2, false, false);

        t0 = *(const v8bf*)(bRow3 + k0); t1 = *(const v8bf*)(bRow3 + k0 + 8);
        b = __builtin_shufflevector(t0, t1, 0,1,2,3,4,5,6,7,8,9,10,11,12,13,14,15);
        c3 = __builtin_amdgcn_wmma_f32_16x16x32_bf16(false, a, false, b, (short)0, c3, false, false);
    }

    // Epilogue.  C/D layout: lane<16 -> (M=r, N=lane); lane>=16 -> (M=8+r, N=lane-16)
    const int rBase = tileM + ((lane >= 16) ? 8 : 0);
    #pragma unroll
    for (int t = 0; t < 4; ++t) {
        v8f cv = (t == 0) ? c0 : (t == 1) ? c1 : (t == 2) ? c2 : c3;
        int col = tileN + t * 16 + (lane & 15);
        float bs = 0.f;
        if (bias)  bs += bias[col];
        if (bias2) bs += bias2[col];
        float sg = 1.f, sb = 0.f, sm = 0.f;
        if (bng) { sg = bng[col] * rsqrtf(bnv[col] + EPSC); sb = bnb[col]; sm = bnm[col]; }
        #pragma unroll
        for (int r = 0; r < 8; ++r) {
            float v = cv[r] + bs;
            if (elu) v = v > 0.f ? v : expm1f(v);
            if (bng) v = (v - sm) * sg + sb;
            int row = rBase + r;
            if (outF) outF[(size_t)row * ldf + col] = v;
            if (outH) outH[(size_t)row * ldh + colOff + col] = (bf16_t)v;
        }
    }
}

// ---------------------------------------------------------------- attention (tiny: VALU)
__global__ __launch_bounds__(64)
void attention_k(const float* __restrict__ Qf, const bf16_t* __restrict__ Kh,
                 const bf16_t* __restrict__ Vh, bf16_t* __restrict__ ctx) {
    __shared__ float qs[DK_];
    __shared__ float sc[W_];
    const int b = blockIdx.x >> 4;
    const int h = blockIdx.x & 15;
    const int t = threadIdx.x;

    qs[t] = Qf[(size_t)b * H_ + h * DK_ + t];
    __syncthreads();
    if (t < W_) {
        const bf16_t* kr = Kh + ((size_t)(b * W_ + t)) * H_ + h * DK_;
        float s = 0.f;
        #pragma unroll
        for (int d = 0; d < DK_; ++d) s += qs[d] * (float)kr[d];
        sc[t] = s * 0.125f;                 // 1/sqrt(64)
    }
    __syncthreads();
    float mx = -1e30f;
    #pragma unroll
    for (int w = 0; w < W_; ++w) mx = fmaxf(mx, sc[w]);
    float den = 0.f;
    #pragma unroll
    for (int w = 0; w < W_; ++w) den += expf(sc[w] - mx);
    float inv = 1.f / den;
    const bf16_t* vcol = Vh + (size_t)b * W_ * H_ + h * DK_ + t;
    float acc = 0.f;
    #pragma unroll
    for (int w = 0; w < W_; ++w)
        acc += expf(sc[w] - mx) * inv * (float)vcol[(size_t)w * H_];
    ctx[(size_t)b * H_ + h * DK_ + t] = (bf16_t)acc;
}

// ---------------------------------------------------------------- LSTM gates + post-BN
__global__ void gates_k(const float* __restrict__ pre, const float* __restrict__ pc,
                        const float* __restrict__ g, const float* __restrict__ bb,
                        const float* __restrict__ m, const float* __restrict__ v,
                        float* __restrict__ out) {
    size_t i = (size_t)blockIdx.x * blockDim.x + threadIdx.x;
    if (i >= (size_t)B_ * H_) return;
    int row = (int)(i / H_), j = (int)(i % H_);
    const float* pr = pre + (size_t)row * N4H_;
    float iv = tanhf(pr[j]);
    float f  = 1.f / (1.f + expf(-pr[H_ + j]));
    float ii = 1.f / (1.f + expf(-pr[2 * H_ + j]));
    float o  = 1.f / (1.f + expf(-pr[3 * H_ + j]));
    float cell = iv * ii + pc[i] * f;
    float e = cell > 0.f ? cell : expm1f(cell);
    float h = o * e;
    out[i] = (h - m[j]) * (g[j] * rsqrtf(v[j] + EPSC)) + bb[j];
}

// ---------------------------------------------------------------- host launch
extern "C" void kernel_launch(void* const* d_in, const int* in_sizes, int n_in,
                              void* d_out, int out_size, void* d_ws, size_t ws_size,
                              hipStream_t stream) {
    (void)in_sizes; (void)n_in; (void)out_size; (void)ws_size;
    const float* x    = (const float*)d_in[0];
    const float* ph   = (const float*)d_in[1];
    const float* pc   = (const float*)d_in[2];
    const float* pcs  = (const float*)d_in[3];
    const float* W_ih = (const float*)d_in[4];
    const float* b_ih = (const float*)d_in[5];
    const float* W_hh = (const float*)d_in[6];
    const float* W_q  = (const float*)d_in[7];
    const float* b_q  = (const float*)d_in[8];
    const float* W_ac = (const float*)d_in[9];
    const float* b_ac = (const float*)d_in[10];
    const float* Wq_a = (const float*)d_in[11];
    const float* bq_a = (const float*)d_in[12];
    const float* Wk_a = (const float*)d_in[13];
    const float* bk_a = (const float*)d_in[14];
    const float* Wv_a = (const float*)d_in[15];
    const float* bv_a = (const float*)d_in[16];
    const float* Wo_a = (const float*)d_in[17];
    const float* bo_a = (const float*)d_in[18];
    const float* bag  = (const float*)d_in[19];
    const float* bab  = (const float*)d_in[20];
    const float* bam  = (const float*)d_in[21];
    const float* bav  = (const float*)d_in[22];
    const float* bpg  = (const float*)d_in[23];
    const float* bpb  = (const float*)d_in[24];
    const float* bpm  = (const float*)d_in[25];
    const float* bpv  = (const float*)d_in[26];

    char* p = (char*)d_ws;
    auto alloc = [&](size_t bytes) -> void* {
        void* r = (void*)p;
        p += (bytes + 255) & ~(size_t)255;
        return r;
    };
    bf16_t* xha  = (bf16_t*)alloc((size_t)B_ * KCAT_ * 2);
    bf16_t* wcat = (bf16_t*)alloc((size_t)N4H_ * KCAT_ * 2);
    bf16_t* wq   = (bf16_t*)alloc((size_t)H_ * KQ_ * 2);
    bf16_t* wqa  = (bf16_t*)alloc((size_t)H_ * H_ * 2);
    bf16_t* wka  = (bf16_t*)alloc((size_t)H_ * KKV_ * 2);
    bf16_t* wva  = (bf16_t*)alloc((size_t)H_ * KKV_ * 2);
    bf16_t* woa  = (bf16_t*)alloc((size_t)H_ * H_ * 2);
    bf16_t* kv   = (bf16_t*)alloc((size_t)B_ * W_ * KKV_ * 2);
    bf16_t* qh   = (bf16_t*)alloc((size_t)B_ * H_ * 2);
    float*  Qf   = (float*) alloc((size_t)B_ * H_ * 4);
    bf16_t* Kh   = (bf16_t*)alloc((size_t)B_ * W_ * H_ * 2);
    bf16_t* Vh   = (bf16_t*)alloc((size_t)B_ * W_ * H_ * 2);
    bf16_t* ctxh = (bf16_t*)alloc((size_t)B_ * H_ * 2);
    float*  pre  = (float*) alloc((size_t)B_ * N4H_ * 4);

    auto cvt = [&](const float* in, bf16_t* out, size_t n) {
        f32_to_bf16_k<<<(unsigned)((n + 255) / 256), 256, 0, stream>>>(in, out, n);
    };
    cvt(W_q,  wq,  (size_t)H_ * KQ_);
    cvt(Wq_a, wqa, (size_t)H_ * H_);
    cvt(Wk_a, wka, (size_t)H_ * KKV_);
    cvt(Wv_a, wva, (size_t)H_ * KKV_);
    cvt(Wo_a, woa, (size_t)H_ * H_);

    {
        size_t n = (size_t)N4H_ * KCAT_;
        build_wcat_k<<<(unsigned)((n + 255) / 256), 256, 0, stream>>>(W_ih, W_hh, W_ac, wcat);
    }
    {
        size_t n = (size_t)B_ * KQ_;
        build_xha_k<<<(unsigned)((n + 255) / 256), 256, 0, stream>>>(x, ph, xha);
    }
    {
        size_t n = (size_t)B_ * W_ * KKV_;
        build_kv_k<<<(unsigned)((n + 255) / 256), 256, 0, stream>>>(pcs, kv);
    }

    auto gemm = [&](const bf16_t* A, int lda, const bf16_t* Wt, int ldw,
                    const float* bias, const float* bias2, int M, int N, int K, int elu,
                    const float* g, const float* bb, const float* m, const float* v,
                    float* outF, int ldf, bf16_t* outH, int ldh, int colOff) {
        dim3 grid((unsigned)(N / 64), (unsigned)(M / 64));
        gemm_bf16_wmma<<<grid, 128, 0, stream>>>(A, lda, Wt, ldw, bias, bias2, M, N, K, elu,
                                                 g, bb, m, v, outF, ldf, outH, ldh, colOff);
    };

    // q = elu([x,h] @ W_q^T + b_q)  -> bf16
    gemm(xha, KCAT_, wq, KQ_, b_q, nullptr, B_, H_, KQ_, 1,
         nullptr, nullptr, nullptr, nullptr, nullptr, 0, qh, H_, 0);
    // Q = q @ Wq_a^T + bq_a  -> f32
    gemm(qh, H_, wqa, H_, bq_a, nullptr, B_, H_, H_, 0,
         nullptr, nullptr, nullptr, nullptr, Qf, H_, nullptr, 0, 0);
    // K = elu(kv @ Wk_a^T + bk_a)  -> bf16   (dominant GEMM)
    gemm(kv, KKV_, wka, KKV_, bk_a, nullptr, B_ * W_, H_, KKV_, 1,
         nullptr, nullptr, nullptr, nullptr, nullptr, 0, Kh, H_, 0);
    // V = elu(kv @ Wv_a^T + bv_a)  -> bf16   (dominant GEMM)
    gemm(kv, KKV_, wva, KKV_, bv_a, nullptr, B_ * W_, H_, KKV_, 1,
         nullptr, nullptr, nullptr, nullptr, nullptr, 0, Vh, H_, 0);

    attention_k<<<B_ * HEADS_, 64, 0, stream>>>(Qf, Kh, Vh, ctxh);

    // attn = BN(elu(ctx @ Wo_a^T + bo_a)) -> bf16 into xha cols [1536:2560]
    gemm(ctxh, H_, woa, H_, bo_a, nullptr, B_, H_, H_, 1,
         bag, bab, bam, bav, nullptr, 0, xha, KCAT_, IN_ + H_);
    // pre = [x,h,attn] @ Wcat^T + b_ih + b_ac -> f32
    gemm(xha, KCAT_, wcat, KCAT_, b_ih, b_ac, B_, N4H_, KCAT_, 0,
         nullptr, nullptr, nullptr, nullptr, pre, N4H_, nullptr, 0, 0);

    {
        size_t n = (size_t)B_ * H_;
        gates_k<<<(unsigned)((n + 255) / 256), 256, 0, stream>>>(pre, pc, bpg, bpb, bpm, bpv,
                                                                 (float*)d_out);
    }
}
